// Encoder_62354335203749
// MI455X (gfx1250) — matrix operand order repaired and checked by reading, not compiled, not run on hardware
//
#include <hip/hip_runtime.h>
#include <hip/hip_bf16.h>

typedef __attribute__((ext_vector_type(16))) _Float16 v16h;
typedef __attribute__((ext_vector_type(8)))  float    v8f;

#define LNUM  6
#define CDIM  256
#define FCDIM 1024
#define HNUM  4
#define WREL  4
#define BNUM  4
#define TDIM  1024
#define KCDIM 64

__device__ inline v8f vzero8() {
    v8f z;
#pragma unroll
    for (int i = 0; i < 8; ++i) z[i] = 0.0f;
    return z;
}

__device__ inline v8f wmma_f16(v16h a, v16h b, v8f c) {
    // D = A(16x32 f16) * B(32x16 f16) + C(16x16 f32)
    return __builtin_amdgcn_wmma_f32_16x16x32_f16(false, a, false, b, (short)0, c, false, false);
}

// Async global->LDS copy (16B per lane), tracked by ASYNCcnt (ISA 08 §4).
__device__ inline void async_ld_b128(unsigned lds_addr, const void* gaddr) {
    asm volatile("global_load_async_to_lds_b128 %0, %1, off"
                 :: "v"(lds_addr), "v"(gaddr) : "memory");
}
__device__ inline void wait_async_le1() { asm volatile("s_wait_asynccnt 0x1" ::: "memory"); }
__device__ inline void wait_async_0()   { asm volatile("s_wait_asynccnt 0x0" ::: "memory"); }

// ---------------------------------------------------------------------------
// Pack a row-major f32 matrix W[M,K] (with arbitrary row/col strides, e.g.
// conv taps) into f16 A-matrix tiles (16Mx32K), tile order mt-major/kt-minor,
// within a tile: [lane][element] with K(e,h) = e + 8h + 8*(e>=8), M = lane&15.
// dst index == flat thread id by construction.
// ---------------------------------------------------------------------------
__global__ void pack_a_kernel(const float* __restrict__ src, _Float16* __restrict__ dst,
                              int K, int rowStride, int colStride, float scale, int total) {
    int tid = blockIdx.x * blockDim.x + threadIdx.x;
    if (tid >= total) return;
    int e    = tid & 15;
    int lane = (tid >> 4) & 31;
    int tile = tid >> 9;
    int Kt = K >> 5;
    int mt = tile / Kt, kt = tile % Kt;
    int h = lane >> 4;
    int m = mt * 16 + (lane & 15);
    int k = kt * 32 + e + 8 * h + ((e >= 8) ? 8 : 0);
    dst[tid] = (_Float16)(src[(size_t)m * rowStride + (size_t)k * colStride] * scale);
}

// ---------------------------------------------------------------------------
// Pack an activation f32 matrix [K,N] (or its transpose-in-memory if srcT=1)
// into f16 B-matrix tiles (32Kx16N), tile order nt-major/kt-minor,
// within a tile: [lane][element] with K(e,h) = e + 8h + 8*(e>=8), N = lane&15.
// Optional column shift (conv taps, zero-padded) and time-mask multiply.
// ---------------------------------------------------------------------------
__global__ void pack_b_kernel(const float* __restrict__ src, _Float16* __restrict__ dst,
                              int K, int N, int shift, const float* __restrict__ mask,
                              long long srcBatchStride, long long dstBatchStride,
                              int srcT, int totalPerBatch) {
    int b = blockIdx.y;
    int tid = blockIdx.x * blockDim.x + threadIdx.x;
    if (tid >= totalPerBatch) return;
    int e    = tid & 15;
    int lane = (tid >> 4) & 31;
    int tile = tid >> 9;
    int Kt = K >> 5;
    int nt = tile / Kt, kt = tile % Kt;
    int h = lane >> 4;
    int n = nt * 16 + (lane & 15);
    int k = kt * 32 + e + 8 * h + ((e >= 8) ? 8 : 0);
    int col = n + shift;
    float v = 0.0f;
    if (col >= 0 && col < N) {
        const float* s = src + (size_t)b * srcBatchStride;
        v = srcT ? s[(size_t)col * K + k] : s[(size_t)k * N + col];
        if (mask) v *= mask[(size_t)b * N + col];   // only used when N==TDIM, batches==BNUM
    }
    dst[(size_t)b * dstBatchStride + tid] = (_Float16)v;
}

// erk -> per-layer B-matrix [K=64(d), N=16(delta, 9 valid)]: 2 k-tiles/layer.
__global__ void pack_erk_kernel(const float* __restrict__ erk, _Float16* __restrict__ dst) {
    int tid = blockIdx.x * blockDim.x + threadIdx.x;
    if (tid >= LNUM * 1024) return;
    int l = tid >> 10;
    int r = tid & 1023;
    int kt = r >> 9;
    int lane = (r >> 4) & 31, e = r & 15;
    int n = lane & 15, h = lane >> 4;
    int d = kt * 32 + e + 8 * h + ((e >= 8) ? 8 : 0);
    float v = (n < 2 * WREL + 1) ? erk[((size_t)l * 9 + n) * KCDIM + d] : 0.0f;
    dst[tid] = (_Float16)v;
}

// erv -> per-layer B-matrix [K=32(delta pad, 9 valid), N=64(d)]: 4 n-tiles/layer.
__global__ void pack_erv_kernel(const float* __restrict__ erv, _Float16* __restrict__ dst) {
    int tid = blockIdx.x * blockDim.x + threadIdx.x;
    if (tid >= LNUM * 2048) return;
    int l = tid >> 11;
    int r = tid & 2047;
    int nt = r >> 9;
    int lane = (r >> 4) & 31, e = r & 15;
    int n = lane & 15, h = lane >> 4;
    int kk = e + 8 * h + ((e >= 8) ? 8 : 0);
    int d = nt * 16 + n;
    float v = (kk < 2 * WREL + 1) ? erv[((size_t)l * 9 + kk) * KCDIM + d] : 0.0f;
    dst[tid] = (_Float16)v;
}

// ---------------------------------------------------------------------------
// Generic WMMA GEMM: out[b][M,N] = sum_taps A_tap[M,K] * B_tap[b][K,N] + bias.
// Block = 256 threads (8 waves); block tile = 128(M) x 64(N); wave = 16 x 64.
// A fragments: direct 32B/lane global loads from pre-packed f16 tiles.
// B tiles: async global->LDS DMA (GLOBAL_LOAD_ASYNC_TO_LDS_B128, ASYNCcnt),
// double-buffered so stage s+1 copies while stage s feeds the WMMAs; the 8
// waves share each 4KB stage instead of re-fetching it 8x from global.
// ---------------------------------------------------------------------------
__global__ void gemm_kernel(const _Float16* __restrict__ Apk, const _Float16* __restrict__ Bpk,
                            const float* __restrict__ bias, float* __restrict__ out,
                            int K, int N, int taps,
                            long long tapStrideA, long long tapStrideB,
                            long long bBatchStride, long long outBatchStride,
                            float biasScale, int relu) {
    int b    = blockIdx.z;
    int wave = threadIdx.x >> 5;
    int lane = threadIdx.x & 31;
    int mt   = blockIdx.x * 8 + wave;
    int nt0  = blockIdx.y * 4;
    int Kt   = K >> 5;
    int stages = taps * Kt;

    __shared__ __align__(16) _Float16 bstage[2][2048];   // 2 x 4KB (4 B-tiles)

    // Per-thread DMA slot: 16 bytes of the 4KB stage (256 threads == 4KB).
    int flat = threadIdx.x * 16;     // byte offset in stage
    int jt   = flat >> 10;           // which of the 4 n-tiles
    int boff = flat & 1023;          // byte offset within that tile

    const _Float16* bbase = Bpk + (size_t)b * bBatchStride;
    unsigned ldsbase = (unsigned)(size_t)(&bstage[0][0]);   // LDS-relative offset

    auto issue = [&](int s, int buf) {
        int tap = s / Kt, kt = s % Kt;
        const char* src = (const char*)(bbase + (size_t)tap * tapStrideB
                                        + (((size_t)(nt0 + jt) * Kt + kt) << 9)) + boff;
        async_ld_b128(ldsbase + (unsigned)(buf * 4096 + flat), src);
    };

    v8f acc[4];
#pragma unroll
    for (int j = 0; j < 4; ++j) acc[j] = vzero8();

    issue(0, 0);
    for (int s = 0; s < stages; ++s) {
        int tap = s / Kt, kt = s % Kt;
        if (s + 1 < stages) { issue(s + 1, (s + 1) & 1); wait_async_le1(); }
        else                { wait_async_0(); }
        __syncthreads();    // stage s visible to all waves

        v16h a = *(const v16h*)(Apk + (size_t)tap * tapStrideA
                                + (((size_t)mt * Kt + kt) << 9) + lane * 16);
        if (s + 1 < stages) {
            int tap1 = (s + 1) / Kt, kt1 = (s + 1) % Kt;
            __builtin_prefetch(Apk + (size_t)tap1 * tapStrideA
                               + (((size_t)mt * Kt + kt1) << 9) + lane * 16, 0, 0);
        }
        const _Float16* bs = &bstage[s & 1][0];
#pragma unroll
        for (int j = 0; j < 4; ++j) {
            v16h bf = *(const v16h*)(bs + j * 512 + lane * 16);
            acc[j] = wmma_f16(a, bf, acc[j]);
        }
        __syncthreads();    // all reads of this buffer done before it is re-filled
    }

    int n = lane & 15, h = lane >> 4;
#pragma unroll
    for (int j = 0; j < 4; ++j) {
        int col = (nt0 + j) * 16 + n;
#pragma unroll
        for (int r = 0; r < 8; ++r) {
            int row = mt * 16 + r + 8 * h;
            float v = acc[j][r] + bias[row] * biasScale;
            if (relu) v = fmaxf(v, 0.0f);
            out[(size_t)b * outBatchStride + (size_t)row * N + col] = v;
        }
    }
}

// ---------------------------------------------------------------------------
// Fused banded-relative attention, flash-style online softmax.
// One wave (32 threads) per (b, head, 16-row query tile).
//   q:    f32 [B,C,T] (already scaled by 1/sqrt(KC) via weight pack)
//   kpk:  f16 B-tiles of k-head slices [K=64(d), N=T], batch = b*H+h
//   vpk:  f16 B-tiles of v-head-transpose [K=T(m), N=64(d)]
//   erkp: f16 B-tiles [64 x 16(delta)]  (this layer)
//   ervp: f16 B-tiles [32(delta) x 64]  (this layer)
// Output attnT f32 [B, T, C] (transposed for coalesced stores).
// ---------------------------------------------------------------------------
__global__ void attn_kernel(const float* __restrict__ q,
                            const _Float16* __restrict__ kpk, const _Float16* __restrict__ vpk,
                            const _Float16* __restrict__ erkp, const _Float16* __restrict__ ervp,
                            const float* __restrict__ xmask, float* __restrict__ attnT) {
    int l0 = blockIdx.x * 16;
    int bh = blockIdx.y;
    int b = bh / HNUM, hd = bh % HNUM;
    int lane = threadIdx.x;
    int nl = lane & 15, hh = lane >> 4;

    __shared__ float    Rlds[16 * 16];
    __shared__ _Float16 pbuf[16 * 32];

    // --- Q fragments (A layout), gathered once per tile ---
    const float* qbase = q + ((size_t)b * CDIM + hd * KCDIM) * TDIM;
    v16h aq0, aq1;
#pragma unroll
    for (int e = 0; e < 16; ++e) {
        int kk = e + 8 * hh + ((e >= 8) ? 8 : 0);
        aq0[e] = (_Float16)qbase[(size_t)kk * TDIM + l0 + nl];
        aq1[e] = (_Float16)qbase[(size_t)(32 + kk) * TDIM + l0 + nl];
    }

    // --- rel-K scores R[l][delta] = Q_tile . erk^T  (one 16x16 result) ---
    {
        v8f racc = vzero8();
        v16h bk0 = *(const v16h*)(erkp + 0 * 512 + lane * 16);
        v16h bk1 = *(const v16h*)(erkp + 1 * 512 + lane * 16);
        racc = wmma_f16(aq0, bk0, racc);
        racc = wmma_f16(aq1, bk1, racc);
#pragma unroll
        for (int r = 0; r < 8; ++r) Rlds[(r + 8 * hh) * 16 + nl] = racc[r];
    }
    __syncthreads();

    float rmask[8];
#pragma unroll
    for (int r = 0; r < 8; ++r) rmask[r] = xmask[(size_t)b * TDIM + l0 + r + 8 * hh];

    v8f O[4];
#pragma unroll
    for (int dt = 0; dt < 4; ++dt) O[dt] = vzero8();
    float runM[8], runS[8];
#pragma unroll
    for (int r = 0; r < 8; ++r) { runM[r] = -1e30f; runS[r] = 0.0f; }

    const _Float16* kbb = kpk + (size_t)bh * (KCDIM * TDIM);
    const _Float16* vtb = vpk + (size_t)bh * (KCDIM * TDIM);

    for (int mc = 0; mc < TDIM / 32; ++mc) {
        int m0 = mc * 32;
        v8f Sf[2];
#pragma unroll
        for (int j = 0; j < 2; ++j) {
            v8f s = vzero8();
            // scores: Q[16x64] . K^T chunk, K packed: nt = m-tile, Kt=2
            const _Float16* t0 = kbb + (((size_t)(m0 / 16 + j) * 2 + 0) << 9) + lane * 16;
            const _Float16* t1 = kbb + (((size_t)(m0 / 16 + j) * 2 + 1) << 9) + lane * 16;
            s = wmma_f16(aq0, *(const v16h*)t0, s);
            s = wmma_f16(aq1, *(const v16h*)t1, s);
            int m = m0 + j * 16 + nl;
            float cm = xmask[(size_t)b * TDIM + m];
#pragma unroll
            for (int r = 0; r < 8; ++r) {
                int ll = r + 8 * hh;
                int delta = m - (l0 + ll);
                float v = s[r];
                if (delta >= -WREL && delta <= WREL) v += Rlds[ll * 16 + delta + WREL];
                if (cm * rmask[r] == 0.0f) v = -1e4f;
                s[r] = v;
            }
            Sf[j] = s;
        }

        // --- online softmax: row max over 16-lane groups, rescale ---
        float sc[8];
#pragma unroll
        for (int r = 0; r < 8; ++r) {
            float v = fmaxf(Sf[0][r], Sf[1][r]);
            v = fmaxf(v, __shfl_xor(v, 1, 32));
            v = fmaxf(v, __shfl_xor(v, 2, 32));
            v = fmaxf(v, __shfl_xor(v, 4, 32));
            v = fmaxf(v, __shfl_xor(v, 8, 32));
            float nm = fmaxf(runM[r], v);
            sc[r] = __expf(runM[r] - nm);
            runM[r] = nm;
            runS[r] *= sc[r];
        }
#pragma unroll
        for (int dt = 0; dt < 4; ++dt)
#pragma unroll
            for (int r = 0; r < 8; ++r) O[dt][r] *= sc[r];

        __syncthreads();   // previous-iteration pbuf readers done
#pragma unroll
        for (int j = 0; j < 2; ++j)
#pragma unroll
            for (int r = 0; r < 8; ++r) {
                float p = __expf(Sf[j][r] - runM[r]);
                Sf[j][r] = p;
                pbuf[(r + 8 * hh) * 32 + j * 16 + nl] = (_Float16)p;
            }
#pragma unroll
        for (int r = 0; r < 8; ++r) {
            float v = Sf[0][r] + Sf[1][r];
            v += __shfl_xor(v, 1, 32);
            v += __shfl_xor(v, 2, 32);
            v += __shfl_xor(v, 4, 32);
            v += __shfl_xor(v, 8, 32);
            runS[r] += v;
        }
        __syncthreads();

        // --- P (A layout) from LDS: two contiguous 8-half runs per lane ---
        v16h pa;
#pragma unroll
        for (int e = 0; e < 8; ++e)  pa[e] = pbuf[nl * 32 + 8 * hh + e];
#pragma unroll
        for (int e = 8; e < 16; ++e) pa[e] = pbuf[nl * 32 + 16 + 8 * hh + (e - 8)];

        // --- O += P . V chunk ---
#pragma unroll
        for (int dt = 0; dt < 4; ++dt) {
            const v16h bv = *(const v16h*)(vtb + (((size_t)dt * (TDIM / 32) + mc) << 9) + lane * 16);
            O[dt] = wmma_f16(pa, bv, O[dt]);
        }

        // --- banded rel-V: only chunks overlapping [l0-W, l0+15+W] ---
        if (m0 + 31 >= l0 - WREL && m0 <= l0 + 15 + WREL) {
            v16h par;
#pragma unroll
            for (int e = 0; e < 16; ++e) {
                int kk = e + 8 * hh + ((e >= 8) ? 8 : 0);
                float v = 0.0f;
                if (kk < 2 * WREL + 1) {
                    int m = l0 + nl + kk - WREL;
                    if (m >= m0 && m < m0 + 32 && m >= 0 && m < TDIM)
                        v = (float)pbuf[nl * 32 + (m - m0)];
                }
                par[e] = (_Float16)v;
            }
#pragma unroll
            for (int dt = 0; dt < 4; ++dt) {
                const v16h bev = *(const v16h*)(ervp + (size_t)dt * 512 + lane * 16);
                O[dt] = wmma_f16(par, bev, O[dt]);
            }
        }
    }

    // --- normalize and store (attnT is [B,T,C] so lanes write contiguously) ---
#pragma unroll
    for (int dt = 0; dt < 4; ++dt)
#pragma unroll
        for (int r = 0; r < 8; ++r) {
            float v = O[dt][r] / runS[r];
            int row = hd * KCDIM + dt * 16 + nl;
            int col = l0 + r + 8 * hh;
            attnT[((size_t)b * TDIM + col) * CDIM + row] = v;
        }
}

// x = LayerNorm_channels(x + y*maskOpt) * g + b ; one block per (t, b), C threads.
__global__ void add_ln_kernel(float* __restrict__ x, const float* __restrict__ y,
                              const float* __restrict__ g, const float* __restrict__ bta,
                              const float* __restrict__ mask) {
    int t = blockIdx.x, b = blockIdx.y, c = threadIdx.x;
    size_t idx = ((size_t)b * CDIM + c) * TDIM + t;
    float ym = y[idx];
    if (mask) ym *= mask[(size_t)b * TDIM + t];
    float v = x[idx] + ym;
    __shared__ float red[CDIM];
    red[c] = v; __syncthreads();
    for (int s = CDIM / 2; s > 0; s >>= 1) { if (c < s) red[c] += red[c + s]; __syncthreads(); }
    float mean = red[0] * (1.0f / CDIM);
    __syncthreads();
    float d = v - mean;
    red[c] = d * d; __syncthreads();
    for (int s = CDIM / 2; s > 0; s >>= 1) { if (c < s) red[c] += red[c + s]; __syncthreads(); }
    float var = red[0] * (1.0f / CDIM);
    x[idx] = d * rsqrtf(var + 1e-5f) * g[c] + bta[c];
}

__global__ void mask_mul_kernel(const float* __restrict__ src, const float* __restrict__ mask,
                                float* __restrict__ dst, long long total) {
    long long i = (long long)blockIdx.x * blockDim.x + threadIdx.x;
    if (i >= total) return;
    int t = (int)(i % TDIM);
    int b = (int)((i / TDIM) / CDIM);
    dst[i] = src[i] * mask[(size_t)b * TDIM + t];
}

extern "C" void kernel_launch(void* const* d_in, const int* in_sizes, int n_in,
                              void* d_out, int out_size, void* d_ws, size_t ws_size,
                              hipStream_t stream) {
    (void)in_sizes; (void)n_in; (void)out_size; (void)ws_size;
    const float* x_in  = (const float*)d_in[0];
    const float* xmask = (const float*)d_in[1];
    const float* wq    = (const float*)d_in[2];
    const float* bq    = (const float*)d_in[3];
    const float* wk    = (const float*)d_in[4];
    const float* bk    = (const float*)d_in[5];
    const float* wv    = (const float*)d_in[6];
    const float* bv    = (const float*)d_in[7];
    const float* wo    = (const float*)d_in[8];
    const float* bo    = (const float*)d_in[9];
    const float* erk   = (const float*)d_in[10];
    const float* erv   = (const float*)d_in[11];
    const float* ln1g  = (const float*)d_in[12];
    const float* ln1b  = (const float*)d_in[13];
    const float* fw1   = (const float*)d_in[14];
    const float* fb1   = (const float*)d_in[15];
    const float* fw2   = (const float*)d_in[16];
    const float* fb2   = (const float*)d_in[17];
    const float* ln2g  = (const float*)d_in[18];
    const float* ln2b  = (const float*)d_in[19];
    float* out = (float*)d_out;

    char* wp = (char*)d_ws;
    auto alloc = [&](size_t bytes) -> char* {
        char* r = wp; wp += (bytes + 255) & ~(size_t)255; return r;
    };
    const size_t NCT = (size_t)BNUM * CDIM * TDIM;   // 1M elems
    const size_t NFT = (size_t)BNUM * FCDIM * TDIM;  // 4M elems
    const size_t CC  = (size_t)CDIM * CDIM;
    const size_t FCC = (size_t)FCDIM * CDIM;

    float* xb    = (float*)alloc(NCT * 4);
    float* qf    = (float*)alloc(NCT * 4);
    float* kf    = (float*)alloc(NCT * 4);
    float* vf    = (float*)alloc(NCT * 4);
    float* attnT = (float*)alloc(NCT * 4);
    float* y1    = (float*)alloc(NFT * 4);
    float* y2    = (float*)alloc(NCT * 4);
    _Float16* wqpk  = (_Float16*)alloc(LNUM * CC * 2);
    _Float16* wkpk  = (_Float16*)alloc(LNUM * CC * 2);
    _Float16* wvpk  = (_Float16*)alloc(LNUM * CC * 2);
    _Float16* wopk  = (_Float16*)alloc(LNUM * CC * 2);
    _Float16* w1pk  = (_Float16*)alloc((size_t)LNUM * 3 * FCC * 2);
    _Float16* w2pk  = (_Float16*)alloc((size_t)LNUM * 3 * FCC * 2);
    _Float16* erkpk = (_Float16*)alloc(LNUM * 1024 * 2);
    _Float16* ervpk = (_Float16*)alloc(LNUM * 2048 * 2);
    _Float16* xpk   = (_Float16*)alloc(NCT * 2);
    _Float16* kpk   = (_Float16*)alloc(NCT * 2);
    _Float16* vpk   = (_Float16*)alloc(NCT * 2);
    _Float16* apk   = (_Float16*)alloc(NCT * 2);
    _Float16* fpk   = (_Float16*)alloc(3 * NCT * 2);
    _Float16* hpk   = (_Float16*)alloc(3 * NFT * 2);

    // ---- pack all weights (every call; deterministic) ----
    for (int i = 0; i < LNUM; ++i) {
        pack_a_kernel<<<CC / 256, 256, 0, stream>>>(wq + i * CC, wqpk + i * CC, CDIM, CDIM, 1, 0.125f, (int)CC);
        pack_a_kernel<<<CC / 256, 256, 0, stream>>>(wk + i * CC, wkpk + i * CC, CDIM, CDIM, 1, 1.0f, (int)CC);
        pack_a_kernel<<<CC / 256, 256, 0, stream>>>(wv + i * CC, wvpk + i * CC, CDIM, CDIM, 1, 1.0f, (int)CC);
        pack_a_kernel<<<CC / 256, 256, 0, stream>>>(wo + i * CC, wopk + i * CC, CDIM, CDIM, 1, 1.0f, (int)CC);
        for (int tap = 0; tap < 3; ++tap) {
            pack_a_kernel<<<FCC / 256, 256, 0, stream>>>(
                fw1 + (size_t)i * FCC * 3 + tap, w1pk + ((size_t)i * 3 + tap) * FCC,
                CDIM, CDIM * 3, 3, 1.0f, (int)FCC);
            pack_a_kernel<<<FCC / 256, 256, 0, stream>>>(
                fw2 + (size_t)i * FCC * 3 + tap, w2pk + ((size_t)i * 3 + tap) * FCC,
                FCDIM, FCDIM * 3, 3, 1.0f, (int)FCC);
        }
    }
    pack_erk_kernel<<<(LNUM * 1024) / 256, 256, 0, stream>>>(erk, erkpk);
    pack_erv_kernel<<<(LNUM * 2048) / 256, 256, 0, stream>>>(erv, ervpk);

    // x = x_in * mask
    mask_mul_kernel<<<(int)(NCT / 256), 256, 0, stream>>>(x_in, xmask, xb, (long long)NCT);

    const long long CT = (long long)CDIM * TDIM;
    const long long FT = (long long)FCDIM * TDIM;
    const long long KT = (long long)KCDIM * TDIM;

    for (int i = 0; i < LNUM; ++i) {
        // --- QKV projections ---
        pack_b_kernel<<<dim3((unsigned)(CT / 256), BNUM), 256, 0, stream>>>(
            xb, xpk, CDIM, TDIM, 0, nullptr, CT, CT, 0, (int)CT);
        gemm_kernel<<<dim3(2, 16, BNUM), 256, 0, stream>>>(
            wqpk + i * CC, xpk, bq + i * CDIM, qf, CDIM, TDIM, 1, 0, 0, CT, CT, 0.125f, 0);
        gemm_kernel<<<dim3(2, 16, BNUM), 256, 0, stream>>>(
            wkpk + i * CC, xpk, bk + i * CDIM, kf, CDIM, TDIM, 1, 0, 0, CT, CT, 1.0f, 0);
        gemm_kernel<<<dim3(2, 16, BNUM), 256, 0, stream>>>(
            wvpk + i * CC, xpk, bv + i * CDIM, vf, CDIM, TDIM, 1, 0, 0, CT, CT, 1.0f, 0);

        // --- attention (fused, banded rel-pos) ---
        pack_b_kernel<<<dim3((unsigned)(KT / 256), BNUM * HNUM), 256, 0, stream>>>(
            kf, kpk, KCDIM, TDIM, 0, nullptr, KT, KT, 0, (int)KT);
        pack_b_kernel<<<dim3((unsigned)(KT / 256), BNUM * HNUM), 256, 0, stream>>>(
            vf, vpk, TDIM, KCDIM, 0, nullptr, KT, KT, 1, (int)KT);
        attn_kernel<<<dim3(TDIM / 16, BNUM * HNUM), 32, 0, stream>>>(
            qf, kpk, vpk, erkpk + (size_t)i * 1024, ervpk + (size_t)i * 2048, xmask, attnT);

        // --- output projection + LN1 ---
        pack_b_kernel<<<dim3((unsigned)(CT / 256), BNUM), 256, 0, stream>>>(
            attnT, apk, CDIM, TDIM, 0, nullptr, CT, CT, 1, (int)CT);
        gemm_kernel<<<dim3(2, 16, BNUM), 256, 0, stream>>>(
            wopk + i * CC, apk, bo + i * CDIM, y2, CDIM, TDIM, 1, 0, 0, CT, CT, 1.0f, 0);
        add_ln_kernel<<<dim3(TDIM, BNUM), CDIM, 0, stream>>>(
            xb, y2, ln1g + i * CDIM, ln1b + i * CDIM, nullptr);

        // --- FFN conv1 (K=3 as 3-tap GEMM) + ReLU ---
        for (int tap = 0; tap < 3; ++tap)
            pack_b_kernel<<<dim3((unsigned)(CT / 256), BNUM), 256, 0, stream>>>(
                xb, fpk + (size_t)tap * NCT, CDIM, TDIM, tap - 1, xmask, CT, CT, 0, (int)CT);
        gemm_kernel<<<dim3(FCDIM / 128, 16, BNUM), 256, 0, stream>>>(
            w1pk + (size_t)i * 3 * FCC, fpk, fb1 + i * FCDIM, y1, CDIM, TDIM, 3,
            (long long)FCC, (long long)NCT, CT, FT, 1.0f, 1);

        // --- FFN conv2 + mask + LN2 ---
        for (int tap = 0; tap < 3; ++tap)
            pack_b_kernel<<<dim3((unsigned)(FT / 256), BNUM), 256, 0, stream>>>(
                y1, hpk + (size_t)tap * NFT, FCDIM, TDIM, tap - 1, xmask, FT, FT, 0, (int)FT);
        gemm_kernel<<<dim3(2, 16, BNUM), 256, 0, stream>>>(
            w2pk + (size_t)i * 3 * FCC, hpk, fb2 + i * CDIM, y2, FCDIM, TDIM, 3,
            (long long)FCC, (long long)NFT, FT, CT, 1.0f, 0);
        add_ln_kernel<<<dim3(TDIM, BNUM), CDIM, 0, stream>>>(
            xb, y2, ln2g + i * CDIM, ln2b + i * CDIM, xmask);
    }

    // out = x * mask
    mask_mul_kernel<<<(int)(NCT / 256), 256, 0, stream>>>(xb, xmask, out, (long long)NCT);
}
